// Mamba2LMHeadModel_30889404793192
// MI455X (gfx1250) — compile-verified, matrix operand order checked by reading, not run on hardware
//
#include <hip/hip_runtime.h>

// ---------------- model constants ----------------
#define SEQ    2048
#define DM     768
#define DI     1536
#define NH     24
#define HD     64
#define DSTATE 128
#define CD     1792          // DI + 2*DSTATE
#define DIP    3352          // 2*DI + 2*DSTATE + NH
#define LDZ    3360          // DIP padded to /16
#define NCHUNK 32
#define CLEN   64
#define VOC    50288
#define NLAYER 4

typedef __attribute__((ext_vector_type(16))) __bf16 bf16x16;
typedef __attribute__((ext_vector_type(8)))  __bf16 bf16x8;
typedef __attribute__((ext_vector_type(8)))  float  f32x8;

union FragU { bf16x16 v; bf16x8 h[2]; };

__device__ __forceinline__ bf16x16 ld_frag(const __bf16* p) {
  FragU u;
  u.h[0] = *(const bf16x8*)(p);
  u.h[1] = *(const bf16x8*)(p + 16);
  return u.v;
}

__device__ __forceinline__ f32x8 wmma_bf16(bf16x16 a, bf16x16 b, f32x8 c) {
  // D = A x B + C, 16x16x32 bf16 -> f32, wave32
  return __builtin_amdgcn_wmma_f32_16x16x32_bf16(false, a, false, b, (short)0, c, false, false);
}

__device__ __forceinline__ f32x8 zero8() {
  f32x8 v;
#pragma unroll
  for (int i = 0; i < 8; i++) v[i] = 0.f;
  return v;
}

// ----- gfx1250 async global->LDS path (guarded; falls back to sync loader) -----
#if defined(__gfx1250__) && __has_builtin(__builtin_amdgcn_global_load_async_to_lds_b128) && \
    __has_builtin(__builtin_amdgcn_s_wait_asynccnt)
#define HAVE_ASYNC 1
#else
#define HAVE_ASYNC 0
#endif

#if HAVE_ASYNC
// builtin pointee type is the GCC-vector 'int __attribute__((vector_size(16)))'
typedef int v4i_ __attribute__((vector_size(16)));
__device__ __forceinline__ void async_b128(void* lds, const void* g) {
  __builtin_amdgcn_global_load_async_to_lds_b128(
      (__attribute__((address_space(1))) v4i_*)(g),
      (__attribute__((address_space(3))) v4i_*)(lds),
      0, 0);
}
#endif

// ---------------- elementwise / small kernels ----------------

__global__ __launch_bounds__(256) void cvt_bf16_kernel(const float* __restrict__ in,
                                                       __bf16* __restrict__ out, long n) {
  long i = (long)blockIdx.x * 256 + threadIdx.x;
  long stride = (long)gridDim.x * 256;
  for (; i < n; i += stride) out[i] = (__bf16)in[i];
}

__global__ __launch_bounds__(256) void embed_kernel(const int* __restrict__ ids,
                                                    const float* __restrict__ emb,
                                                    float* __restrict__ x) {
  int l = blockIdx.x;
  long r = (long)ids[l];
  for (int i = threadIdx.x; i < DM; i += 256)
    x[(long)l * DM + i] = emb[r * DM + i];
}

__global__ __launch_bounds__(256) void rmsnorm_kernel(const float* __restrict__ x,
                                                      const float* __restrict__ w,
                                                      __bf16* __restrict__ out, int ncols) {
  const int row = blockIdx.x, tid = threadIdx.x;
  const float* xr = x + (long)row * ncols;
  float ss = 0.f;
  for (int i = tid; i < ncols; i += 256) { float v = xr[i]; ss += v * v; }
  __shared__ float red[256];
  red[tid] = ss; __syncthreads();
  for (int s = 128; s > 0; s >>= 1) { if (tid < s) red[tid] += red[tid + s]; __syncthreads(); }
  float rs = rsqrtf(red[0] / (float)ncols + 1e-5f);
  for (int i = tid; i < ncols; i += 256)
    out[(long)row * ncols + i] = (__bf16)(xr[i] * rs * w[i]);
}

__global__ __launch_bounds__(256) void gated_rmsnorm_kernel(const float* __restrict__ y,
                                                            const float* __restrict__ z,
                                                            const float* __restrict__ w,
                                                            __bf16* __restrict__ out) {
  const int row = blockIdx.x, tid = threadIdx.x;
  float v[6];
  float ss = 0.f;
#pragma unroll
  for (int j = 0; j < 6; j++) {
    int i = tid + j * 256;
    float zz = z[(long)row * LDZ + i];
    float g = y[(long)row * DI + i] * (zz / (1.f + __expf(-zz)));
    v[j] = g; ss += g * g;
  }
  __shared__ float red[256];
  red[tid] = ss; __syncthreads();
  for (int s = 128; s > 0; s >>= 1) { if (tid < s) red[tid] += red[tid + s]; __syncthreads(); }
  float rs = rsqrtf(red[0] / (float)DI + 1e-5f);
#pragma unroll
  for (int j = 0; j < 6; j++) {
    int i = tid + j * 256;
    out[(long)row * DI + i] = (__bf16)(v[j] * rs * w[i]);
  }
}

__global__ __launch_bounds__(256) void conv_silu_kernel(const float* __restrict__ zx,
                                                        const float* __restrict__ cw,
                                                        const float* __restrict__ cb,
                                                        float* __restrict__ out) {
  int idx = blockIdx.x * 256 + threadIdx.x;
  if (idx >= SEQ * CD) return;
  int l = idx / CD, ch = idx % CD;
  float acc = cb[ch];
#pragma unroll
  for (int j = 0; j < 4; j++) {
    int ls = l - 3 + j;
    if (ls >= 0) acc += zx[(long)ls * LDZ + DI + ch] * cw[ch * 4 + j];
  }
  out[(long)l * CD + ch] = acc / (1.f + __expf(-acc));   // silu
}

__global__ __launch_bounds__(256) void dt_softplus_kernel(const float* __restrict__ zx,
                                                          const float* __restrict__ dtb,
                                                          float* __restrict__ out) {
  int idx = blockIdx.x * 256 + threadIdx.x;
  if (idx >= SEQ * NH) return;
  int l = idx / NH, h = idx % NH;
  float v = zx[(long)l * LDZ + DI + CD + h] + dtb[h];
  out[idx] = (v > 20.f) ? v : log1pf(__expf(v));
}

// Sequential inter-chunk state recurrence: S_0=0; Sprev[c]=S; S = S*exp(Asum_c) + Schunk_c
__global__ __launch_bounds__(256) void ssd_scan_kernel(const float* __restrict__ Schunk,
                                                       const float* __restrict__ gAcum,
                                                       float* __restrict__ Sprev) {
  const int h = blockIdx.x;
  const int tid = threadIdx.x;
  float s[32];
#pragma unroll
  for (int j = 0; j < 32; j++) s[j] = 0.f;
  for (int c = 0; c < NCHUNK; c++) {
    long base = ((long)c * NH + h) * (CLEN * DSTATE);
    float dA = __expf(gAcum[((long)c * NH + h) * CLEN + (CLEN - 1)]);
#pragma unroll
    for (int j = 0; j < 32; j++) {
      int idx = tid + j * 256;
      Sprev[base + idx] = s[j];
      s[j] = s[j] * dA + Schunk[base + idx];
    }
  }
}

// ---------------- WMMA NT GEMM: C[M,N](f32,ldc) = A[M,K](bf16) * B[N,K]^T (bf16) ----------------
#define GBM 128
#define GBN 128
#define GBK 32
#define GLD (GBK + 8)    // padded LDS row stride (elems); 80B rows keep b128 alignment

__global__ __launch_bounds__(256) void gemm_nt_bf16(const __bf16* __restrict__ A,
                                                    const __bf16* __restrict__ B,
                                                    float* __restrict__ C,
                                                    int M, int N, int K, int ldc, int accum) {
#if HAVE_ASYNC
  __shared__ __align__(16) __bf16 lA[2][GBM * GLD];
  __shared__ __align__(16) __bf16 lB[2][GBN * GLD];
#else
  __shared__ __align__(16) __bf16 lA[1][GBM * GLD];
  __shared__ __align__(16) __bf16 lB[1][GBN * GLD];
#endif
  const int tid = threadIdx.x;
  const int lane = tid & 31, w = tid >> 5;
  const int half = lane >> 4, l16 = lane & 15;
  const long bm = (long)blockIdx.y * GBM;
  const long bn = (long)blockIdx.x * GBN;
  const int wm = (w & 3) * 32;   // 4 waves along M, 2 tiles each
  const int wn = (w >> 2) * 64;  // 2 waves along N, 4 tiles each

  // per-thread loader coordinates: 512 chunks of 8 bf16 per matrix per K-step
  const int r0 = tid >> 2;            // 0..63
  const int c8 = (tid & 3) * 8;
  const long ar0 = bm + r0, ar1 = ar0 + 64;
  // clamp out-of-range B rows: they only feed guarded (never stored) columns
  long br0 = bn + r0;      if (br0 > (long)N - 1) br0 = (long)N - 1;
  long br1 = bn + r0 + 64; if (br1 > (long)N - 1) br1 = (long)N - 1;

  f32x8 acc[2][4];
#pragma unroll
  for (int i = 0; i < 2; i++)
#pragma unroll
    for (int j = 0; j < 4; j++) acc[i][j] = zero8();

  const int nk = K / GBK;

#if HAVE_ASYNC
  // ---- async double-buffered pipeline: ASYNCcnt-tracked direct global->LDS ----
  {
    async_b128(&lA[0][r0 * GLD + c8],        A + ar0 * (long)K + c8);
    async_b128(&lA[0][(r0 + 64) * GLD + c8], A + ar1 * (long)K + c8);
    async_b128(&lB[0][r0 * GLD + c8],        B + br0 * (long)K + c8);
    async_b128(&lB[0][(r0 + 64) * GLD + c8], B + br1 * (long)K + c8);
  }
  for (int t = 0; t < nk; t++) {
    __builtin_amdgcn_s_wait_asynccnt(0);
    __syncthreads();
    const int cur = t & 1;
    if (t + 1 < nk) {
      const int nxt = cur ^ 1;
      const long k1 = (long)(t + 1) * GBK;
      async_b128(&lA[nxt][r0 * GLD + c8],        A + ar0 * (long)K + k1 + c8);
      async_b128(&lA[nxt][(r0 + 64) * GLD + c8], A + ar1 * (long)K + k1 + c8);
      async_b128(&lB[nxt][r0 * GLD + c8],        B + br0 * (long)K + k1 + c8);
      async_b128(&lB[nxt][(r0 + 64) * GLD + c8], B + br1 * (long)K + k1 + c8);
    }
    bf16x16 af[2], bfr[4];
#pragma unroll
    for (int i = 0; i < 2; i++)
      af[i] = ld_frag(&lA[cur][(wm + i * 16 + l16) * GLD + half * 8]);
#pragma unroll
    for (int j = 0; j < 4; j++)
      bfr[j] = ld_frag(&lB[cur][(wn + j * 16 + l16) * GLD + half * 8]);
#pragma unroll
    for (int i = 0; i < 2; i++)
#pragma unroll
      for (int j = 0; j < 4; j++)
        acc[i][j] = wmma_bf16(af[i], bfr[j], acc[i][j]);
  }
#else
  // ---- fallback: synchronous loader (known-good) ----
  for (int t = 0; t < nk; t++) {
    const long k0 = (long)t * GBK;
    {
      const __bf16* ga = A + ar0 * (long)K + k0 + c8;
      *(bf16x8*)&lA[0][r0 * GLD + c8] = *(const bf16x8*)ga;
      __builtin_prefetch(ga + GBK, 0, 1);
      const __bf16* ga1 = A + ar1 * (long)K + k0 + c8;
      *(bf16x8*)&lA[0][(r0 + 64) * GLD + c8] = *(const bf16x8*)ga1;
      *(bf16x8*)&lB[0][r0 * GLD + c8] = *(const bf16x8*)(B + br0 * (long)K + k0 + c8);
      *(bf16x8*)&lB[0][(r0 + 64) * GLD + c8] = *(const bf16x8*)(B + br1 * (long)K + k0 + c8);
    }
    __syncthreads();
    bf16x16 af[2], bfr[4];
#pragma unroll
    for (int i = 0; i < 2; i++)
      af[i] = ld_frag(&lA[0][(wm + i * 16 + l16) * GLD + half * 8]);
#pragma unroll
    for (int j = 0; j < 4; j++)
      bfr[j] = ld_frag(&lB[0][(wn + j * 16 + l16) * GLD + half * 8]);
#pragma unroll
    for (int i = 0; i < 2; i++)
#pragma unroll
      for (int j = 0; j < 4; j++)
        acc[i][j] = wmma_bf16(af[i], bfr[j], acc[i][j]);
    __syncthreads();
  }
#endif

#pragma unroll
  for (int i = 0; i < 2; i++) {
#pragma unroll
    for (int j = 0; j < 4; j++) {
      long col = bn + wn + j * 16 + l16;
      if (col < N) {
        long rbase = bm + wm + i * 16 + half * 8;
#pragma unroll
        for (int e = 0; e < 8; e++) {
          long off = (rbase + e) * (long)ldc + col;
          float v = acc[i][j][e];
          if (accum) C[off] += v; else C[off] = v;
        }
      }
    }
  }
}

// ---------------- SSD intra-chunk kernel (one block per chunk x head) ----------------
// Computes Acum -> global, Y_diag (+D skip) -> y, chunk state -> Schunk. All matmuls WMMA bf16.
__global__ __launch_bounds__(256) void ssd_chunk_kernel(const float* __restrict__ xBC,
                                                        const float* __restrict__ dt,
                                                        const float* __restrict__ A_log,
                                                        const float* __restrict__ Dp,
                                                        float* __restrict__ y,
                                                        float* __restrict__ gAcum,
                                                        float* __restrict__ Schunk) {
  const int c = blockIdx.x;   // chunk
  const int h = blockIdx.y;   // head
  const int tid = threadIdx.x;
  const int lane = tid & 31, w = tid >> 5;
  const int half = lane >> 4, l16 = lane & 15;
  const long bl = (long)c * CLEN;

  // phase-aliased LDS
  __shared__ __align__(16) __bf16 sP1[9216];   // sC (64x128, ld136) -> sBT (128x64, ld72)
  __shared__ __align__(16) __bf16 sP2[8704];   // sB (64x128, ld136) -> sM (64x64, ld136)
  __shared__ __align__(16) float  sP3[4608];   // sG (64x64 f32, ld72) -> sXT (bf16 64x64, ld72)
  __shared__ float sA[CLEN], sDt[CLEN], sAc[CLEN];

  __bf16* sC  = sP1;
  __bf16* sB  = sP2;
  float*  sG  = sP3;
  __bf16* sM  = sP2;
  __bf16* sXT = (__bf16*)sP3;
  __bf16* sBT = sP1;

  // Phase A: A*dt, dt, B, C
  if (tid < CLEN) {
    float dtv = dt[(bl + tid) * NH + h];
    sDt[tid] = dtv;
    sA[tid] = -__expf(A_log[h]) * dtv;
  }
  for (int i = 0; i < 32; i++) {
    int idx = i * 256 + tid;            // 0..8191
    int l = idx >> 7, n = idx & 127;
    sB[l * 136 + n] = (__bf16)xBC[(bl + l) * (long)CD + DI + n];
    sC[l * 136 + n] = (__bf16)xBC[(bl + l) * (long)CD + DI + DSTATE + n];
  }
  __syncthreads();
  if (tid == 0) {
    float run = 0.f;
    for (int l = 0; l < CLEN; l++) { run += sA[l]; sAc[l] = run; }
  }
  __syncthreads();
  if (tid < CLEN) gAcum[((long)c * NH + h) * CLEN + tid] = sAc[tid];

  // Phase B: G = C * B^T   (64x64, K=128)
#pragma unroll
  for (int t = 2 * w; t <= 2 * w + 1; t++) {
    const int tm = (t & 3) * 16, tn = (t >> 2) * 16;
    f32x8 acc = zero8();
#pragma unroll
    for (int k = 0; k < 128; k += 32) {
      bf16x16 a = ld_frag(&sC[(tm + l16) * 136 + k + half * 8]);
      bf16x16 b = ld_frag(&sB[(tn + l16) * 136 + k + half * 8]);
      acc = wmma_bf16(a, b, acc);
    }
#pragma unroll
    for (int e = 0; e < 8; e++)
      sG[(tm + half * 8 + e) * 72 + tn + l16] = acc[e];
  }
  __syncthreads();

  // Phase C: M[l][s] = (s<=l) ? G[l][s]*exp(Ac[l]-Ac[s]) : 0
  for (int i = 0; i < 16; i++) {
    int idx = i * 256 + tid;            // 0..4095
    int l = idx >> 6, s = idx & 63;
    float v = (s <= l) ? sG[l * 72 + s] * __expf(sAc[l] - sAc[s]) : 0.f;
    sM[l * 136 + s] = (__bf16)v;
  }
  __syncthreads();

  // Phase D: XT[p][s] = x[s,h,p] * dt[s]
  for (int i = 0; i < 16; i++) {
    int idx = i * 256 + tid;
    int p = idx >> 6, s = idx & 63;
    float xv = xBC[(bl + s) * (long)CD + h * HD + p] * sDt[s];
    sXT[p * 72 + s] = (__bf16)xv;
  }
  __syncthreads();

  // Phase E: Y_diag = M @ xs  (64x64, K=64), add D-skip, store y
  const float dcoef = Dp[h];
#pragma unroll
  for (int t = 2 * w; t <= 2 * w + 1; t++) {
    const int tm = (t & 3) * 16, tn = (t >> 2) * 16;
    f32x8 acc = zero8();
#pragma unroll
    for (int k = 0; k < 64; k += 32) {
      bf16x16 a = ld_frag(&sM[(tm + l16) * 136 + k + half * 8]);
      bf16x16 b = ld_frag(&sXT[(tn + l16) * 72 + k + half * 8]);
      acc = wmma_bf16(a, b, acc);
    }
    const int pcol = tn + l16;
#pragma unroll
    for (int e = 0; e < 8; e++) {
      int l = tm + half * 8 + e;
      float xv = xBC[(bl + l) * (long)CD + h * HD + pcol];
      y[(bl + l) * (long)DI + h * HD + pcol] = acc[e] + dcoef * xv;
    }
  }
  __syncthreads();

  // Phase F: XT *= decay(s) ; BT[n][s] = B[s][n]
  {
    float alast = sAc[CLEN - 1];
    for (int i = 0; i < 16; i++) {
      int idx = i * 256 + tid;
      int p = idx >> 6, s = idx & 63;
      float v = (float)sXT[p * 72 + s] * __expf(alast - sAc[s]);
      sXT[p * 72 + s] = (__bf16)v;
    }
    for (int i = 0; i < 32; i++) {
      int idx = i * 256 + tid;          // 0..8191
      int n = idx >> 6, s = idx & 63;
      sBT[n * 72 + s] = (__bf16)xBC[(bl + s) * (long)CD + DI + n];
    }
  }
  __syncthreads();

  // Phase G: Schunk[p][n] = (xs*decay)^T @ B  (64x128, K=64)
  long base = ((long)c * NH + h) * (CLEN * DSTATE);
#pragma unroll
  for (int i = 0; i < 4; i++) {
    int t = w + 8 * i;                   // 32 tiles total
    const int tm = (t & 3) * 16;         // p
    const int tn = (t >> 2) * 16;        // n
    f32x8 acc = zero8();
#pragma unroll
    for (int k = 0; k < 64; k += 32) {
      bf16x16 a = ld_frag(&sXT[(tm + l16) * 72 + k + half * 8]);
      bf16x16 b = ld_frag(&sBT[(tn + l16) * 72 + k + half * 8]);
      acc = wmma_bf16(a, b, acc);
    }
#pragma unroll
    for (int e = 0; e < 8; e++) {
      int p = tm + half * 8 + e;
      Schunk[base + p * DSTATE + tn + l16] = acc[e];
    }
  }
}

// ---------------- SSD inter-chunk output: Y_off = exp(Acum) * (C @ Sprev^T) ----------------
__global__ __launch_bounds__(256) void ssd_yoff_kernel(const float* __restrict__ xBC,
                                                       const float* __restrict__ Sprev,
                                                       const float* __restrict__ gAcum,
                                                       float* __restrict__ y) {
  const int c = blockIdx.x, h = blockIdx.y;
  const int tid = threadIdx.x, lane = tid & 31, w = tid >> 5;
  const int half = lane >> 4, l16 = lane & 15;
  const long bl = (long)c * CLEN;
  __shared__ __align__(16) __bf16 sC[64 * 136];
  __shared__ __align__(16) __bf16 sS[64 * 136];
  __shared__ float sE[CLEN];
  if (tid < CLEN) sE[tid] = __expf(gAcum[((long)c * NH + h) * CLEN + tid]);
  long sbase = ((long)c * NH + h) * (CLEN * DSTATE);
  for (int i = 0; i < 32; i++) {
    int idx = i * 256 + tid;
    int r = idx >> 7, n = idx & 127;
    sC[r * 136 + n] = (__bf16)xBC[(bl + r) * (long)CD + DI + DSTATE + n];
    sS[r * 136 + n] = (__bf16)Sprev[sbase + r * DSTATE + n];   // r == p
  }
  __syncthreads();
#pragma unroll
  for (int t = 2 * w; t <= 2 * w + 1; t++) {
    const int tm = (t & 3) * 16, tn = (t >> 2) * 16;
    f32x8 acc = zero8();
#pragma unroll
    for (int k = 0; k < 128; k += 32) {
      bf16x16 a = ld_frag(&sC[(tm + l16) * 136 + k + half * 8]);
      bf16x16 b = ld_frag(&sS[(tn + l16) * 136 + k + half * 8]);
      acc = wmma_bf16(a, b, acc);
    }
    const int pcol = tn + l16;
#pragma unroll
    for (int e = 0; e < 8; e++) {
      int l = tm + half * 8 + e;
      y[(bl + l) * (long)DI + h * HD + pcol] += sE[l] * acc[e];
    }
  }
}

// ---------------- host launch ----------------
extern "C" void kernel_launch(void* const* d_in, const int* in_sizes, int n_in,
                              void* d_out, int out_size, void* d_ws, size_t ws_size,
                              hipStream_t stream) {
  const int*   ids          = (const int*)d_in[0];
  const float* emb          = (const float*)d_in[1];
  const float* in_proj_w    = (const float*)d_in[2];
  const float* conv_w       = (const float*)d_in[3];
  const float* conv_b       = (const float*)d_in[4];
  const float* dt_bias      = (const float*)d_in[5];
  const float* A_log        = (const float*)d_in[6];
  const float* Dp           = (const float*)d_in[7];
  const float* mixer_norm_w = (const float*)d_in[8];
  const float* out_proj_w   = (const float*)d_in[9];
  const float* layer_norm_w = (const float*)d_in[10];
  const float* norm_f_w     = (const float*)d_in[11];

  char* wsp = (char*)d_ws;
  auto take = [&](size_t bytes) -> void* {
    void* p = (void*)wsp;
    wsp += (bytes + 255) & ~(size_t)255;
    return p;
  };

  float*  x      = (float*) take((size_t)SEQ * DM * 4);
  __bf16* xnbf   = (__bf16*)take((size_t)SEQ * DM * 2);
  __bf16* embbf  = (__bf16*)take((size_t)VOC * DM * 2);
  __bf16* inwbf  = (__bf16*)take((size_t)NLAYER * DIP * DM * 2);
  __bf16* outwbf = (__bf16*)take((size_t)NLAYER * DM * DI * 2);
  float*  zx     = (float*) take((size_t)SEQ * LDZ * 4);
  float*  xbc    = (float*) take((size_t)SEQ * CD * 4);
  float*  dtb    = (float*) take((size_t)SEQ * NH * 4);
  float*  ybuf   = (float*) take((size_t)SEQ * DI * 4);
  __bf16* ybf    = (__bf16*)take((size_t)SEQ * DI * 2);
  float*  acum   = (float*) take((size_t)NCHUNK * NH * CLEN * 4);
  float*  schunk = (float*) take((size_t)NCHUNK * NH * CLEN * DSTATE * 4);
  float*  sprev  = (float*) take((size_t)NCHUNK * NH * CLEN * DSTATE * 4);

  // one-time bf16 conversions
  cvt_bf16_kernel<<<4096, 256, 0, stream>>>(emb, embbf, (long)VOC * DM);
  cvt_bf16_kernel<<<4096, 256, 0, stream>>>(in_proj_w, inwbf, (long)NLAYER * DIP * DM);
  cvt_bf16_kernel<<<2048, 256, 0, stream>>>(out_proj_w, outwbf, (long)NLAYER * DM * DI);

  embed_kernel<<<SEQ, 256, 0, stream>>>(ids, emb, x);

  for (int i = 0; i < NLAYER; i++) {
    rmsnorm_kernel<<<SEQ, 256, 0, stream>>>(x, layer_norm_w + i * DM, xnbf, DM);
    gemm_nt_bf16<<<dim3((DIP + GBN - 1) / GBN, SEQ / GBM), 256, 0, stream>>>(
        xnbf, inwbf + (size_t)i * DIP * DM, zx, SEQ, DIP, DM, LDZ, 0);
    conv_silu_kernel<<<(SEQ * CD + 255) / 256, 256, 0, stream>>>(
        zx, conv_w + (size_t)i * CD * 4, conv_b + (size_t)i * CD, xbc);
    dt_softplus_kernel<<<(SEQ * NH + 255) / 256, 256, 0, stream>>>(
        zx, dt_bias + i * NH, dtb);
    ssd_chunk_kernel<<<dim3(NCHUNK, NH), 256, 0, stream>>>(
        xbc, dtb, A_log + i * NH, Dp + i * NH, ybuf, acum, schunk);
    ssd_scan_kernel<<<NH, 256, 0, stream>>>(schunk, acum, sprev);
    ssd_yoff_kernel<<<dim3(NCHUNK, NH), 256, 0, stream>>>(xbc, sprev, acum, ybuf);
    gated_rmsnorm_kernel<<<SEQ, 256, 0, stream>>>(ybuf, zx, mixer_norm_w + i * DI, ybf);
    gemm_nt_bf16<<<dim3(DM / GBN, SEQ / GBM), 256, 0, stream>>>(
        ybf, outwbf + (size_t)i * DM * DI, x, SEQ, DM, DI, DM, 1);  // accumulate residual
  }

  rmsnorm_kernel<<<SEQ, 256, 0, stream>>>(x, norm_f_w, xnbf, DM);
  gemm_nt_bf16<<<dim3((VOC + GBN - 1) / GBN, SEQ / GBM), 256, 0, stream>>>(
      xnbf, embbf, (float*)d_out, SEQ, VOC, DM, VOC, 0);
}
